// Decoder_24223615550272
// MI455X (gfx1250) — matrix operand order repaired and checked by reading, not compiled, not run on hardware
//
#include <hip/hip_runtime.h>
#include <hip/hip_bf16.h>
#include <math.h>

#define BATCH  64
#define EMBED  512
#define HID    1024
#define ATTN   1024
#define MAXLEN 1024
#define VOCAB  50257
#define XDIM   (EMBED + HID)   // 1536

typedef __attribute__((ext_vector_type(16))) __bf16 v16bf;
typedef __attribute__((ext_vector_type(8)))  float  v8f;

union BF16x16 {
  v16bf v;
  unsigned short h[16];
  unsigned int   u[8];
};

// ---------- cheap truncation split: x ~= hi + lo (each bf16), err ~2^-16 ----------
// hi = top 16 bits of x; lo = top 16 bits of (x - hi). Packs pairwise with and/or/shift.
__device__ __forceinline__ void split16(const float* v, BF16x16& hi, BF16x16& lo) {
#pragma unroll
  for (int i = 0; i < 8; ++i) {
    unsigned int u0 = __float_as_uint(v[2 * i + 0]);
    unsigned int u1 = __float_as_uint(v[2 * i + 1]);
    unsigned int h0 = u0 & 0xffff0000u;
    unsigned int h1 = u1 & 0xffff0000u;
    hi.u[i] = (u0 >> 16) | h1;
    unsigned int l0 = __float_as_uint(v[2 * i + 0] - __uint_as_float(h0));
    unsigned int l1 = __float_as_uint(v[2 * i + 1] - __uint_as_float(h1));
    lo.u[i] = (l0 >> 16) | (l1 & 0xffff0000u);
  }
}

// A-fragment (16x32 bf16, contiguous k order, k = 16*(lane/16) + e) from row-major f32.
// Caller passes rowp = &A[m*lda + kbase + (lane>=16 ? 16 : 0)].
// frag elements e=0..15 -> rowp[0..15].
__device__ __forceinline__ void load_afrag_f32(const float* rowp, BF16x16& hi, BF16x16& lo) {
  float4 q0 = *(const float4*)(rowp + 0);
  float4 q1 = *(const float4*)(rowp + 4);
  float4 q2 = *(const float4*)(rowp + 8);
  float4 q3 = *(const float4*)(rowp + 12);
  float v[16] = {q0.x,q0.y,q0.z,q0.w, q1.x,q1.y,q1.z,q1.w,
                 q2.x,q2.y,q2.z,q2.w, q3.x,q3.y,q3.z,q3.w};
  split16(v, hi, lo);
}

// 3-product split-bf16 accumulate: (ahi+alo)(bhi+blo) ~ ahi*bhi + ahi*blo + alo*bhi
__device__ __forceinline__ v8f wmma3(const BF16x16& ahi, const BF16x16& alo,
                                     const BF16x16& bhi, const BF16x16& blo, v8f c) {
  c = __builtin_amdgcn_wmma_f32_16x16x32_bf16(false, ahi.v, false, bhi.v, (short)0, c, false, false);
  c = __builtin_amdgcn_wmma_f32_16x16x32_bf16(false, ahi.v, false, blo.v, (short)0, c, false, false);
  c = __builtin_amdgcn_wmma_f32_16x16x32_bf16(false, alo.v, false, bhi.v, (short)0, c, false, false);
  return c;
}

// ---------------- w1_e pre-pack: f32 -> {hi16|lo16} u32 ----------------
__global__ __launch_bounds__(256)
void pack_w1e_kernel(const float* __restrict__ w_att1, unsigned int* __restrict__ pk) {
  int i = blockIdx.x * 256 + threadIdx.x;        // covers HID*ATTN = 1M
  float f = w_att1[i];                           // w1_e = first HID rows of w_att1
  unsigned int u = __float_as_uint(f);
  unsigned int hbits = u & 0xffff0000u;
  unsigned int l = __float_as_uint(f - __uint_as_float(hbits)) >> 16;
  pk[i] = hbits | l;
}

// ------------- dproj = dec @ w1_d  (B is [K,N] row-major) -------------
__global__ __launch_bounds__(256)
void gemm_bkn_kernel(const float* __restrict__ A, int lda,
                     const float* __restrict__ B, int ldb, int K,
                     int Ntiles, float* __restrict__ C, int ldc) {
  int wt = blockIdx.x * 8 + (threadIdx.x >> 5);
  if (wt >= 4 * Ntiles) return;
  int mt = wt & 3, nt = wt >> 2;
  int lane = threadIdx.x & 31, ln = lane & 15, lhi = lane >> 4;
  int n = nt * 16 + ln;
  v8f acc = {0.f,0.f,0.f,0.f,0.f,0.f,0.f,0.f};
#pragma unroll 1
  for (int kb = 0; kb < K; kb += 32) {
    BF16x16 ahi, alo, bhi, blo;
    load_afrag_f32(A + (size_t)(mt * 16 + ln) * lda + kb + lhi * 16, ahi, alo);
    float bv[16];
    const float* bp = B + (size_t)(kb + lhi * 16) * ldb + n;
#pragma unroll
    for (int e = 0; e < 16; ++e) bv[e] = bp[(size_t)e * ldb];
    __builtin_prefetch(bp + (size_t)32 * ldb, 0, 3);   // next K-block, B stream
    split16(bv, bhi, blo);
    acc = wmma3(ahi, alo, bhi, blo, acc);
  }
#pragma unroll
  for (int r = 0; r < 8; ++r)
    C[(size_t)(mt * 16 + lhi * 8 + r) * ldc + n] = acc[r];
}

// ---- G = A @ W^T + bias  (W row-major [N,K]; per-lane contiguous) ----
__global__ __launch_bounds__(256)
void gemm_bnk_kernel(const float* __restrict__ A, int lda,
                     const float* __restrict__ W, int K,
                     const float* __restrict__ bias, int Nreal, int Ntiles,
                     float* __restrict__ C, int ldc) {
  int wt = blockIdx.x * 8 + (threadIdx.x >> 5);
  if (wt >= 4 * Ntiles) return;
  int mt = wt & 3, nt = wt >> 2;
  int lane = threadIdx.x & 31, ln = lane & 15, lhi = lane >> 4;
  int n = nt * 16 + ln;
  bool nv = (n < Nreal);
  v8f acc = {0.f,0.f,0.f,0.f,0.f,0.f,0.f,0.f};
#pragma unroll 1
  for (int kb = 0; kb < K; kb += 32) {
    BF16x16 ahi, alo, bhi, blo;
    load_afrag_f32(A + (size_t)(mt * 16 + ln) * lda + kb + lhi * 16, ahi, alo);
    if (nv) {
      const float* rp = W + (size_t)n * K + kb + lhi * 16;   // 16 contiguous k
      float4 q0 = *(const float4*)(rp + 0);
      float4 q1 = *(const float4*)(rp + 4);
      float4 q2 = *(const float4*)(rp + 8);
      float4 q3 = *(const float4*)(rp + 12);
      __builtin_prefetch(rp + 32, 0, 3);                     // next K-block of this row
      float bv[16] = {q0.x,q0.y,q0.z,q0.w, q1.x,q1.y,q1.z,q1.w,
                      q2.x,q2.y,q2.z,q2.w, q3.x,q3.y,q3.z,q3.w};
      split16(bv, bhi, blo);
    } else {
#pragma unroll
      for (int e = 0; e < 8; ++e) { bhi.u[e] = 0u; blo.u[e] = 0u; }
    }
    acc = wmma3(ahi, alo, bhi, blo, acc);    // EXEC reconverged here
  }
  if (nv) {
    float bb = bias ? bias[n] : 0.f;
#pragma unroll
    for (int r = 0; r < 8; ++r)
      C[(size_t)(mt * 16 + lhi * 8 + r) * ldc + n] = acc[r] + bb;
  }
}

// ------------- attention scores: e[t,b] = w_att2 . tanh(enc[t] @ w1_e + dproj) -------------
// One workgroup per t. enc[t] (64x1024 f32) staged in LDS as bf16 hi/lo planes (256 KB).
__global__ __launch_bounds__(256)
void attn_kernel(const float* __restrict__ enc, const unsigned int* __restrict__ w1e_pk,
                 const float* __restrict__ dproj, const float* __restrict__ att2,
                 float* __restrict__ e_out) {
  extern __shared__ unsigned short smem[];
  unsigned short* plane_hi = smem;                         // [64][1024] bf16
  unsigned short* plane_lo = smem + BATCH * HID;           // [64][1024] bf16
  float* partial = (float*)(smem + 2 * BATCH * HID);       // [8 waves][64 b]

  const int t = blockIdx.x;
  const int tid = threadIdx.x;
  const int wave = tid >> 5, lane = tid & 31;
  const int ln = lane & 15, lhi = lane >> 4;

  // Stage + convert enc[t] into LDS hi/lo bf16 planes (truncation split).
  const float* encT = enc + (size_t)t * (BATCH * HID);
#pragma unroll 4
  for (int i = tid; i < BATCH * HID; i += 256) {
    float f = encT[i];
    unsigned int u = __float_as_uint(f);
    unsigned int hbits = u & 0xffff0000u;
    plane_hi[i] = (unsigned short)(u >> 16);
    plane_lo[i] = (unsigned short)(__float_as_uint(f - __uint_as_float(hbits)) >> 16);
  }
  for (int i = lane; i < 64; i += 32) partial[wave * 64 + i] = 0.f;
  __syncthreads();

  const v8f vzero = {0.f,0.f,0.f,0.f,0.f,0.f,0.f,0.f};

  // Each wave: 8 N-tiles, each done for all 4 M-tiles (B read once per WG).
  for (int it = 0; it < 8; ++it) {
    const int nt = wave + it * 8;
    const int n  = nt * 16 + ln;
    v8f acc[4];
    acc[0] = vzero; acc[1] = vzero; acc[2] = vzero; acc[3] = vzero;

#pragma unroll 1
    for (int kb = 0; kb < HID; kb += 32) {
      // B fragment: packed u32 carries hi & lo planes; pairwise unpack (2 ops/word/plane)
      unsigned int w[16];
      const unsigned int* bp = w1e_pk + (size_t)(kb + lhi * 16) * ATTN + n;
#pragma unroll
      for (int e = 0; e < 16; ++e) w[e] = bp[(size_t)e * ATTN];
      __builtin_prefetch(bp + (size_t)32 * ATTN, 0, 3);     // next K-block, B stream
      BF16x16 bhi, blo;
#pragma unroll
      for (int i = 0; i < 8; ++i) {
        unsigned int w0 = w[2 * i], w1 = w[2 * i + 1];
        bhi.u[i] = (w0 >> 16) | (w1 & 0xffff0000u);
        blo.u[i] = (w0 & 0xffffu) | (w1 << 16);
      }
#pragma unroll
      for (int mt = 0; mt < 4; ++mt) {
        int base = (mt * 16 + ln) * HID + kb + lhi * 16;    // 16B aligned
        BF16x16 ahi, alo;
        uint4 qh0 = *(const uint4*)(plane_hi + base);
        uint4 qh1 = *(const uint4*)(plane_hi + base + 8);
        ahi.u[0] = qh0.x; ahi.u[1] = qh0.y; ahi.u[2] = qh0.z; ahi.u[3] = qh0.w;
        ahi.u[4] = qh1.x; ahi.u[5] = qh1.y; ahi.u[6] = qh1.z; ahi.u[7] = qh1.w;
        uint4 ql0 = *(const uint4*)(plane_lo + base);
        uint4 ql1 = *(const uint4*)(plane_lo + base + 8);
        alo.u[0] = ql0.x; alo.u[1] = ql0.y; alo.u[2] = ql0.z; alo.u[3] = ql0.w;
        alo.u[4] = ql1.x; alo.u[5] = ql1.y; alo.u[6] = ql1.z; alo.u[7] = ql1.w;
        acc[mt] = wmma3(ahi, alo, bhi, blo, acc[mt]);
      }
    }

    // Fused epilogue: tanh, scale by w_att2[n], reduce over the 16 n's of this tile.
    const float a2 = att2[n];
#pragma unroll
    for (int mt = 0; mt < 4; ++mt) {
#pragma unroll
      for (int r = 0; r < 8; ++r) {
        int b = mt * 16 + lhi * 8 + r;                 // C layout: M = r + 8*lhi
        float v = tanhf(acc[mt][r] + dproj[b * ATTN + n]) * a2;
        v += __shfl_xor(v, 1);
        v += __shfl_xor(v, 2);
        v += __shfl_xor(v, 4);
        v += __shfl_xor(v, 8);                         // stays within 16-lane half
        if (ln == 0) partial[wave * 64 + b] += v;      // deterministic, wave-private
      }
    }
  }
  __syncthreads();
  if (tid < 64) {
    float s = 0.f;
#pragma unroll
    for (int w = 0; w < 8; ++w) s += partial[w * 64 + tid];
    e_out[t * BATCH + tid] = s;
  }
}

// ---------------- softmax over t (axis 0), per batch column ----------------
__global__ void softmax_kernel(const float* __restrict__ e, float* __restrict__ alpha) {
  int b = threadIdx.x;                       // 64 threads
  float m = -1e30f;
  for (int t = 0; t < MAXLEN; ++t) m = fmaxf(m, e[t * BATCH + b]);
  float s = 0.f;
  for (int t = 0; t < MAXLEN; ++t) {
    float ex = expf(e[t * BATCH + b] - m);
    alpha[t * BATCH + b] = ex;
    s += ex;
  }
  float inv = 1.f / s;
  for (int t = 0; t < MAXLEN; ++t) alpha[t * BATCH + b] *= inv;
}

// -------- x = [embedding[cur], sum_t alpha[t,b]*enc[t,b,:]] --------
__global__ __launch_bounds__(256)
void build_x_kernel(const int* __restrict__ cur, const float* __restrict__ emb_tbl,
                    const float* __restrict__ alpha, const float* __restrict__ enc,
                    float* __restrict__ x) {
  int b = blockIdx.y;
  int j = blockIdx.x * 256 + threadIdx.x;    // 0..1535
  if (j < EMBED) {
    int tok = cur[b];
    x[(size_t)b * XDIM + j] = emb_tbl[(size_t)tok * EMBED + j];
  } else {
    int k = j - EMBED;
    float s = 0.f;
    for (int t = 0; t < MAXLEN; ++t)
      s += alpha[t * BATCH + b] * enc[((size_t)t * BATCH + b) * HID + k];
    x[(size_t)b * XDIM + j] = s;
  }
}

// ---------------- GRU gate fusion ----------------
__global__ __launch_bounds__(256)
void gru_gate_kernel(const float* __restrict__ gi, const float* __restrict__ gh,
                     const float* __restrict__ hprev, float* __restrict__ hout) {
  int idx = blockIdx.x * 256 + threadIdx.x;  // 64*1024
  int b = idx >> 10, j = idx & 1023;
  const float* gib = gi + (size_t)b * 3 * HID;
  const float* ghb = gh + (size_t)b * 3 * HID;
  float ir = gib[j], iz = gib[HID + j], in_ = gib[2 * HID + j];
  float hr = ghb[j], hz = ghb[HID + j], hn = ghb[2 * HID + j];
  float r = 1.f / (1.f + expf(-(ir + hr)));
  float z = 1.f / (1.f + expf(-(iz + hz)));
  float nn = tanhf(in_ + r * hn);
  hout[idx] = (1.f - z) * nn + z * hprev[idx];
}

extern "C" void kernel_launch(void* const* d_in, const int* in_sizes, int n_in,
                              void* d_out, int out_size, void* d_ws, size_t ws_size,
                              hipStream_t stream) {
  (void)in_sizes; (void)n_in; (void)out_size; (void)ws_size;
  const int*   cur    = (const int*)d_in[0];
  const float* state  = (const float*)d_in[1];   // (2,64,1024)
  const float* enc    = (const float*)d_in[2];   // (1024,64,1024)
  const float* embed  = (const float*)d_in[3];
  const float* w_att1 = (const float*)d_in[4];   // (2048,1024)
  const float* w_att2 = (const float*)d_in[5];   // (1024,1)
  const float* w_ih0  = (const float*)d_in[6];   // (3072,1536)
  const float* w_hh0  = (const float*)d_in[7];   // (3072,1024)
  const float* b_ih0  = (const float*)d_in[8];
  const float* b_hh0  = (const float*)d_in[9];
  const float* w_ih1  = (const float*)d_in[10];  // (3072,1024)
  const float* w_hh1  = (const float*)d_in[11];  // (3072,1024)
  const float* b_ih1  = (const float*)d_in[12];
  const float* b_hh1  = (const float*)d_in[13];
  const float* w_out  = (const float*)d_in[14];  // (50257,1024)
  const float* b_out  = (const float*)d_in[15];

  float* out = (float*)d_out;                       // (64,50257)
  float* ns  = out + (size_t)BATCH * VOCAB;         // new_state (2,64,1024)
  float* h0  = ns;
  float* h1  = ns + BATCH * HID;

  float* ws     = (float*)d_ws;
  float* dproj  = ws;                               //  64K f32
  float* e_buf  = ws + 65536;                       //  64K
  float* alpha  = ws + 131072;                      //  64K
  float* xbuf   = ws + 196608;                      //  96K (64x1536)
  float* gi_buf = ws + 294912;                      // 192K (64x3072)
  float* gh_buf = ws + 491520;                      // 192K
  unsigned int* w1e_pk = (unsigned int*)(ws + 688128);  // 1M u32 (4 MB)

  // 1) pack w1_e into {hi,lo} bf16 words (L2-resident, reused 1024x)
  pack_w1e_kernel<<<(HID * ATTN) / 256, 256, 0, stream>>>(w_att1, w1e_pk);

  // 2) dproj = state[1] @ w1_d   (64x1024x1024)
  gemm_bkn_kernel<<<32, 256, 0, stream>>>(state + BATCH * HID, HID,
                                          w_att1 + (size_t)HID * ATTN, ATTN, HID,
                                          ATTN / 16, dproj, ATTN);

  // 3) attention scores (the 137 GFLOP GEMM), fused tanh + w_att2 reduction
  size_t attn_smem = (size_t)2 * BATCH * HID * sizeof(unsigned short)
                   + 8 * 64 * sizeof(float);        // 256 KB + 2 KB (<=320 KB WGP LDS)
  hipFuncSetAttribute((const void*)attn_kernel,
                      hipFuncAttributeMaxDynamicSharedMemorySize, (int)attn_smem);
  attn_kernel<<<MAXLEN, 256, attn_smem, stream>>>(enc, w1e_pk, dproj, w_att2, e_buf);

  // 4) softmax over t, per b
  softmax_kernel<<<1, 64, 0, stream>>>(e_buf, alpha);

  // 5) x = [emb | context]
  build_x_kernel<<<dim3(XDIM / 256, BATCH), 256, 0, stream>>>(cur, embed, alpha, enc, xbuf);

  // 6) GRU layer 0
  gemm_bnk_kernel<<<96, 256, 0, stream>>>(xbuf, XDIM, w_ih0, XDIM, b_ih0,
                                          3 * HID, (3 * HID) / 16, gi_buf, 3 * HID);
  gemm_bnk_kernel<<<96, 256, 0, stream>>>(state, HID, w_hh0, HID, b_hh0,
                                          3 * HID, (3 * HID) / 16, gh_buf, 3 * HID);
  gru_gate_kernel<<<(BATCH * HID) / 256, 256, 0, stream>>>(gi_buf, gh_buf, state, h0);

  // 7) GRU layer 1
  gemm_bnk_kernel<<<96, 256, 0, stream>>>(h0, HID, w_ih1, HID, b_ih1,
                                          3 * HID, (3 * HID) / 16, gi_buf, 3 * HID);
  gemm_bnk_kernel<<<96, 256, 0, stream>>>(state + BATCH * HID, HID, w_hh1, HID, b_hh1,
                                          3 * HID, (3 * HID) / 16, gh_buf, 3 * HID);
  gru_gate_kernel<<<(BATCH * HID) / 256, 256, 0, stream>>>(gi_buf, gh_buf,
                                                           state + BATCH * HID, h1);

  // 8) logits = h1 @ w_out.T + b_out  (N=50257, guarded tiles; w_out read once: ~206 MB)
  int nt_out = (VOCAB + 15) / 16;                 // 3142
  int blocks_out = (4 * nt_out + 7) / 8;          // 1571
  gemm_bnk_kernel<<<blocks_out, 256, 0, stream>>>(h1, HID, w_out, HID, b_out,
                                                  VOCAB, nt_out, out, VOCAB);
}
